// AttentionDecoder_1125281432229
// MI455X (gfx1250) — matrix operand order repaired and checked by reading, not compiled, not run on hardware
//
#include <hip/hip_runtime.h>
#include <hip/hip_bf16.h>

typedef __attribute__((ext_vector_type(16))) __bf16 bf16x16;
typedef __attribute__((ext_vector_type(8)))  __bf16 bf16x8;
typedef __attribute__((ext_vector_type(8)))  float  fx8;

namespace {
constexpr int Tn = 4096, Bn = 64, Hn = 256, En = 256, Vn = 50257;
constexpr int K2 = 2 * Hn;            // 512 (attention K, context width)
constexpr int KC = 2 * Hn + En;       // 768 (comb / logits K)
constexpr int NOT_ = (Vn + 15) / 16;  // 3142 output col tiles for logits
constexpr int W2F_ELEMS = 16 * 16 * 32 * 16;  // [kt][nt][lane][j] bf16 frags
}

__device__ __forceinline__ bf16x16 pack16(float4 u0, float4 u1, float4 u2, float4 u3) {
  bf16x16 a;
  a[0]=(__bf16)u0.x; a[1]=(__bf16)u0.y; a[2]=(__bf16)u0.z; a[3]=(__bf16)u0.w;
  a[4]=(__bf16)u1.x; a[5]=(__bf16)u1.y; a[6]=(__bf16)u1.z; a[7]=(__bf16)u1.w;
  a[8]=(__bf16)u2.x; a[9]=(__bf16)u2.y; a[10]=(__bf16)u2.z; a[11]=(__bf16)u2.w;
  a[12]=(__bf16)u3.x; a[13]=(__bf16)u3.y; a[14]=(__bf16)u3.z; a[15]=(__bf16)u3.w;
  return a;
}

// ---------- setup kernels ----------
__global__ void k_init(const float* __restrict__ vvec, float* __restrict__ scal,
                       float* __restrict__ s_acc) {
  __shared__ float red[8];
  int tid = threadIdx.x;
  float a = fabsf(vvec[tid]);                 // Hn == blockDim == 256
#pragma unroll
  for (int m = 1; m < 32; m <<= 1) a += __shfl_xor(a, m, 32);
  if ((tid & 31) == 0) red[tid >> 5] = a;
  __syncthreads();
  if (tid == 0) {
    float s = 0.f;
    for (int k = 0; k < 8; ++k) s += red[k];
    scal[0] = s;                              // Msum = sum(|v|) >= max score
  }
  if (tid < Bn) s_acc[tid] = 0.0f;
}

__global__ void k_zero(float* __restrict__ p) {
  p[blockIdx.x * 256 + threadIdx.x] = 0.0f;
}

// pre[b][n] = b_attn[n] + sum_h h0[b][h] * W_attn[n][h]
__global__ void k_pre(const float* __restrict__ hid, const float* __restrict__ Wattn,
                      const float* __restrict__ battn, float* __restrict__ pre) {
  int gid = blockIdx.x * 256 + threadIdx.x;   // 16384
  int b = gid >> 8, n = gid & 255;
  float s = battn[n];
  const float* hb = hid + b * Hn;
  const float* wr = Wattn + (size_t)n * (3 * Hn);
  for (int h = 0; h < Hn; ++h) s += hb[h] * wr[h];
  pre[gid] = s;
}

__global__ void k_emb(const int* __restrict__ ids, const float* __restrict__ emb,
                      float* __restrict__ embd) {
  int gid = blockIdx.x * 256 + threadIdx.x;   // 16384
  int b = gid >> 8, e = gid & 255;
  embd[gid] = emb[(size_t)ids[b] * En + e];
}

// W_attn[:, H:3H] -> bf16 B-fragment-major: [kt][nt][lane][j], B[k][n]=W_attn[n][H+k]
__global__ void k_w2frag(const float* __restrict__ Wattn, __bf16* __restrict__ w2f) {
  int i = blockIdx.x * 256 + threadIdx.x;     // 131072
  int j = i & 15, lane = (i >> 4) & 31, nt = (i >> 9) & 15, kt = i >> 13;
  int n = nt * 16 + (lane & 15);
  int k = kt * 32 + (lane >> 4) * 8 + (j & 7) + (j >> 3) * 16;
  w2f[i] = (__bf16)Wattn[(size_t)n * (3 * Hn) + Hn + k];
}

// ---------- fused attention: energies GEMM + tanh + dot(v) + shifted-softmax + context ----------
__global__ void __launch_bounds__(256)
k_attn(const float* __restrict__ enc, const float* __restrict__ pre,
       const __bf16* __restrict__ w2f, const float* __restrict__ vvec,
       const float* __restrict__ scal, float* __restrict__ ctx_acc,
       float* __restrict__ s_acc) {
  __shared__ float ctx_lds[16 * K2];          // 32 KB: 16 b x 512 d partial context
  __shared__ float s_lds[16];
  const int tid = threadIdx.x;
  const int lane = tid & 31, wid = tid >> 5;
  const int l16 = lane & 15, half = lane >> 4;
  const int bg = blockIdx.x;                  // b-group 0..3 (16 b each)
  const int t0 = blockIdx.y * 64;             // 64 timesteps per workgroup

  for (int i = tid; i < 16 * K2; i += 256) ctx_lds[i] = 0.0f;
  if (tid < 16) s_lds[tid] = 0.0f;
  __syncthreads();

  const float Ms = scal[0];
  float vlane[16];
#pragma unroll
  for (int nt = 0; nt < 16; ++nt) vlane[nt] = vvec[nt * 16 + l16];

  for (int tt = wid; tt < 64; tt += 8) {      // each wave: 8 timesteps
    const int t = t0 + tt;
    const float* arow = enc + (size_t)(t * Bn + bg * 16 + l16) * K2;
    if (tt + 8 < 64) __builtin_prefetch(arow + (size_t)8 * Bn * K2, 0, 0);

    fx8 acc[16] = {};
#pragma unroll 1
    for (int kt = 0; kt < 16; ++kt) {
      // A fragment: lane -> row l16, K = kt*32 + half*8 + {0..7, 16..23}
      const float4* p = (const float4*)(arow + kt * 32 + half * 8);
      bf16x16 a = pack16(p[0], p[1], p[4], p[5]);
      const bf16x16* bp = (const bf16x16*)(w2f + (size_t)(kt * 16) * 32 * 16 + (size_t)lane * 16);
#pragma unroll
      for (int nt = 0; nt < 16; ++nt) {
        bf16x16 bfrag = bp[nt * 32];          // nt stride = 32 lanes * 16 elems
        acc[nt] = __builtin_amdgcn_wmma_f32_16x16x32_bf16(
            false, a, false, bfrag, (short)0, acc[nt], false, false);
      }
    }
    // energies = tanh(pre + gemm); score partials over this lane's column residues
    float sp[8] = {};
#pragma unroll
    for (int nt = 0; nt < 16; ++nt) {
#pragma unroll
      for (int e = 0; e < 8; ++e) {
        float g = tanhf(acc[nt][e] +
                        pre[(size_t)(bg * 16 + e + 8 * half) * Hn + nt * 16 + l16]);
        sp[e] += g * vlane[nt];
      }
    }
    // reduce across the 16 column lanes of each half (xor masks stay within halves)
#pragma unroll
    for (int m = 1; m < 16; m <<= 1) {
#pragma unroll
      for (int e = 0; e < 8; ++e) sp[e] += __shfl_xor(sp[e], m, 32);
    }
    float w8[8];
#pragma unroll
    for (int e = 0; e < 8; ++e) w8[e] = __expf(sp[e] - Ms);   // shifted softmax weight
    if (l16 == 0) {
#pragma unroll
      for (int e = 0; e < 8; ++e) atomicAdd(&s_lds[8 * half + e], w8[e]);
    }
    float wall[16];                            // broadcast all 16 row weights
#pragma unroll
    for (int e = 0; e < 8; ++e) {
      wall[e]     = __shfl(w8[e], 0, 32);
      wall[e + 8] = __shfl(w8[e], 16, 32);
    }
    // context partial: lane covers d in [lane*16, lane*16+16) (f32 enc, full precision)
    const float* crow = enc + (size_t)(t * Bn + bg * 16) * K2 + lane * 16;
#pragma unroll
    for (int r = 0; r < 16; ++r) {
      const float4* ep = (const float4*)(crow + (size_t)r * K2);
      float q[16];
      *(float4*)(q)      = ep[0];
      *(float4*)(q + 4)  = ep[1];
      *(float4*)(q + 8)  = ep[2];
      *(float4*)(q + 12) = ep[3];
      float wr = wall[r];
      float* cb = &ctx_lds[r * K2 + lane * 16];
#pragma unroll
      for (int jj = 0; jj < 16; ++jj) atomicAdd(cb + jj, wr * q[jj]);
    }
  }
  __syncthreads();
  for (int i = tid; i < 16 * K2; i += 256)
    atomicAdd(&ctx_acc[(size_t)(bg * 16 + (i >> 9)) * K2 + (i & 511)], ctx_lds[i]);
  if (tid < 16) atomicAdd(&s_acc[bg * 16 + tid], s_lds[tid]);
}

// ---------- combine / comb / GRU ----------
__global__ void k_combine(const float* __restrict__ ctx_acc, const float* __restrict__ s_acc,
                          float* __restrict__ ctx, __bf16* __restrict__ act) {
  int gid = blockIdx.x * 256 + threadIdx.x;   // 32768
  int b = gid >> 9, d = gid & 511;
  float c = ctx_acc[gid] / s_acc[b];
  ctx[gid] = c;
  act[(size_t)b * KC + Hn + d] = (__bf16)c;   // logits activation cols 256..767
}

__global__ void k_comb(const float* __restrict__ ctx, const float* __restrict__ embd,
                       const float* __restrict__ Wcomb, const float* __restrict__ bcomb,
                       float* __restrict__ comb) {
  int gid = blockIdx.x * 256 + threadIdx.x;   // 16384
  int b = gid >> 8, o = gid & 255;
  float s = bcomb[o];
  const float* wr = Wcomb + (size_t)o * KC;
  const float* cb = ctx + b * K2;
  const float* eb = embd + b * En;
  for (int d = 0; d < K2; ++d) s += cb[d] * wr[d];
  for (int e = 0; e < En; ++e) s += eb[e] * wr[K2 + e];
  comb[gid] = s;
}

__global__ void k_gru(const float* __restrict__ comb, const float* __restrict__ hid,
                      const float* __restrict__ Wih, const float* __restrict__ Whh,
                      const float* __restrict__ bih, const float* __restrict__ bhh,
                      __bf16* __restrict__ act, float* __restrict__ outh) {
  int gid = blockIdx.x * 256 + threadIdx.x;   // 16384
  int b = gid >> 8, j = gid & 255;
  const float* xb = comb + b * Hn;
  const float* hb = hid + b * Hn;
  float xr = bih[j], xz = bih[Hn + j], xn = bih[2 * Hn + j];
  float hr = bhh[j], hz = bhh[Hn + j], hn = bhh[2 * Hn + j];
  const float* wr0 = Wih + (size_t)j * Hn;
  const float* wr1 = Wih + (size_t)(Hn + j) * Hn;
  const float* wr2 = Wih + (size_t)(2 * Hn + j) * Hn;
  const float* vr0 = Whh + (size_t)j * Hn;
  const float* vr1 = Whh + (size_t)(Hn + j) * Hn;
  const float* vr2 = Whh + (size_t)(2 * Hn + j) * Hn;
  for (int h = 0; h < Hn; ++h) {
    float xv = xb[h], hv = hb[h];
    xr += xv * wr0[h]; xz += xv * wr1[h]; xn += xv * wr2[h];
    hr += hv * vr0[h]; hz += hv * vr1[h]; hn += hv * vr2[h];
  }
  float r = 1.f / (1.f + __expf(-(xr + hr)));
  float z = 1.f / (1.f + __expf(-(xz + hz)));
  float n = tanhf(xn + r * hn);
  float hnew = (1.f - z) * n + z * hb[j];
  act[(size_t)b * KC + j] = (__bf16)hnew;     // logits activation cols 0..255
  outh[gid] = hnew;                           // d_out hidden output (1,B,H)
}

// ---------- logits GEMM (bf16 WMMA, W_out streamed f32->bf16) ----------
__global__ void __launch_bounds__(256)
k_logits(const __bf16* __restrict__ act, const float* __restrict__ Wout,
         const float* __restrict__ bout, float* __restrict__ out) {
  const int lane = threadIdx.x & 31, wid = threadIdx.x >> 5;
  const int ot = blockIdx.x * 8 + wid;        // one 16-col tile per wave
  if (ot >= NOT_) return;
  const int l16 = lane & 15, half = lane >> 4;
  const int o = ot * 16 + l16;
  const int oc = o < Vn ? o : Vn - 1;

  fx8 acc[4] = {};
#pragma unroll 1
  for (int kt = 0; kt < 24; ++kt) {           // K = 768 = 24 * 32
    // B fragment: lane -> col o, K = kt*32 + half*16 + j  (B[k][o] = W_out[o][k])
    const float4* wp = (const float4*)(Wout + (size_t)oc * KC + kt * 32 + half * 16);
    bf16x16 b = pack16(wp[0], wp[1], wp[2], wp[3]);
#pragma unroll
    for (int mb = 0; mb < 4; ++mb) {          // 64 batch rows = 4 row tiles
      const __bf16* ap = act + (size_t)(mb * 16 + l16) * KC + kt * 32 + half * 16;
      bf16x8 lo = *(const bf16x8*)ap;
      bf16x8 hi = *(const bf16x8*)(ap + 8);
      bf16x16 a;
#pragma unroll
      for (int jj = 0; jj < 8; ++jj) { a[jj] = lo[jj]; a[jj + 8] = hi[jj]; }
      acc[mb] = __builtin_amdgcn_wmma_f32_16x16x32_bf16(
          false, a, false, b, (short)0, acc[mb], false, false);
    }
  }
  if (o < Vn) {
    const float bias = bout[o];
#pragma unroll
    for (int mb = 0; mb < 4; ++mb) {
#pragma unroll
      for (int e = 0; e < 8; ++e)
        out[(size_t)(mb * 16 + e + 8 * half) * Vn + o] = acc[mb][e] + bias;
    }
  }
}

// ---------- log-softmax over V per row ----------
__global__ void k_lse(const float* __restrict__ out, float* __restrict__ lse) {
  __shared__ float red[8];
  __shared__ float bval;
  const int b = blockIdx.x, tid = threadIdx.x;
  const float* row = out + (size_t)b * Vn;
  float m = -3.4e38f;
  for (int i = tid; i < Vn; i += 256) m = fmaxf(m, row[i]);
#pragma unroll
  for (int s = 1; s < 32; s <<= 1) m = fmaxf(m, __shfl_xor(m, s, 32));
  if ((tid & 31) == 0) red[tid >> 5] = m;
  __syncthreads();
  if (tid == 0) {
    float mm = red[0];
    for (int k = 1; k < 8; ++k) mm = fmaxf(mm, red[k]);
    bval = mm;
  }
  __syncthreads();
  const float M = bval;
  float s = 0.f;
  for (int i = tid; i < Vn; i += 256) s += __expf(row[i] - M);
#pragma unroll
  for (int k = 1; k < 32; k <<= 1) s += __shfl_xor(s, k, 32);
  if ((tid & 31) == 0) red[tid >> 5] = s;
  __syncthreads();
  if (tid == 0) {
    float ss = 0.f;
    for (int k = 0; k < 8; ++k) ss += red[k];
    lse[b] = M + __logf(ss);
  }
}

__global__ void k_logp(float* __restrict__ out, const float* __restrict__ lse) {
  int gid = blockIdx.x * 256 + threadIdx.x;
  if (gid < Bn * Vn) {
    int b = gid / Vn;
    out[gid] = out[gid] - lse[b];
  }
}

// ---------- launch ----------
extern "C" void kernel_launch(void* const* d_in, const int* in_sizes, int n_in,
                              void* d_out, int out_size, void* d_ws, size_t ws_size,
                              hipStream_t stream) {
  const int*   ids   = (const int*)d_in[0];
  const float* hid   = (const float*)d_in[1];
  const float* enc   = (const float*)d_in[2];
  const float* emb   = (const float*)d_in[3];
  const float* Wattn = (const float*)d_in[4];
  const float* battn = (const float*)d_in[5];
  const float* vvec  = (const float*)d_in[6];
  const float* Wcomb = (const float*)d_in[7];
  const float* bcomb = (const float*)d_in[8];
  const float* Wih   = (const float*)d_in[9];
  const float* Whh   = (const float*)d_in[10];
  const float* bih   = (const float*)d_in[11];
  const float* bhh   = (const float*)d_in[12];
  const float* Wout  = (const float*)d_in[13];
  const float* bout  = (const float*)d_in[14];
  float* out = (float*)d_out;

  // workspace layout (~0.82 MB), all chunks 32B-aligned
  float*  scal    = (float*)d_ws;             // [0] = Msum
  float*  s_acc   = scal + 64;                // 64
  float*  pre     = s_acc + 64;               // 16384
  float*  embd    = pre + Bn * Hn;            // 16384
  float*  ctx_acc = embd + Bn * En;           // 32768
  float*  ctx     = ctx_acc + Bn * K2;        // 32768
  float*  comb    = ctx + Bn * K2;            // 16384
  float*  lse     = comb + Bn * Hn;           // 64
  __bf16* w2f     = (__bf16*)(lse + 64);      // 131072 bf16
  __bf16* act     = w2f + W2F_ELEMS;          // 64*768 bf16

  k_init  <<<1, 256, 0, stream>>>(vvec, scal, s_acc);
  k_zero  <<<(Bn * K2) / 256, 256, 0, stream>>>(ctx_acc);
  k_pre   <<<(Bn * Hn) / 256, 256, 0, stream>>>(hid, Wattn, battn, pre);
  k_emb   <<<(Bn * En) / 256, 256, 0, stream>>>(ids, emb, embd);
  k_w2frag<<<W2F_ELEMS / 256, 256, 0, stream>>>(Wattn, w2f);
  k_attn  <<<dim3(4, Tn / 64), 256, 0, stream>>>(enc, pre, w2f, vvec, scal, ctx_acc, s_acc);
  k_combine<<<(Bn * K2) / 256, 256, 0, stream>>>(ctx_acc, s_acc, ctx, act);
  k_comb  <<<(Bn * Hn) / 256, 256, 0, stream>>>(ctx, embd, Wcomb, bcomb, comb);
  k_gru   <<<(Bn * Hn) / 256, 256, 0, stream>>>(comb, hid, Wih, Whh, bih, bhh, act,
                                                out + (size_t)Bn * Vn);
  k_logits<<<(NOT_ + 7) / 8, 256, 0, stream>>>(act, Wout, bout, out);
  k_lse   <<<Bn, 256, 0, stream>>>(out, lse);
  k_logp  <<<(Bn * Vn + 255) / 256, 256, 0, stream>>>(out, lse);
}
